// ValueNetwork_75548474737176
// MI455X (gfx1250) — matrix operand order repaired and checked by reading, not compiled, not run on hardware
//
#include <hip/hip_runtime.h>

typedef _Float16 v16h __attribute__((ext_vector_type(16)));
typedef float    v8f  __attribute__((ext_vector_type(8)));

#define WAVES 8
#define BE    8            // batch elements per block
#define RR    (BE * 20)    // 160 rows (batch*nodes) per block
#define RT    (RR / 16)    // 10 row tiles

// Packed-weight offsets in d_ws (in halves). Each WMMA B-tile = 512 halves.
enum {
  OW_M1W0  = 0,                     // NT=10, KT=1
  OW_M1W1  = OW_M1W0  + 10*1*512,   // NT=7,  KT=5
  OW_M2W0  = OW_M1W1  + 7*5*512,    // NT=7,  KT=4
  OW_M2W1  = OW_M2W0  + 7*4*512,    // NT=4,  KT=4
  OW_AW0LO = OW_M2W1  + 4*4*512,
  OW_AW0HI = OW_AW0LO + 7*4*512,
  OW_AW1   = OW_AW0HI + 7*4*512,
  OW_M3W0  = OW_AW1   + 7*4*512,    // NT=10, KT=2
  OW_M3W1  = OW_M3W0  + 10*2*512,   // NT=7,  KT=5
  OW_M3W2  = OW_M3W1  + 7*5*512,    // NT=7,  KT=4
  OW_END   = OW_M3W2  + 7*4*512     // 131072 halves = 256 KiB
};

// Bias offsets inside the LDS bias pool (floats, zero-padded to tile widths).
enum {
  BO_M1B0 = 0,            // 160
  BO_M1B1 = BO_M1B0 + 160,  // 112
  BO_M2B0 = BO_M1B1 + 112,
  BO_M2B1 = BO_M2B0 + 112,  // 64
  BO_AB0  = BO_M2B1 + 64,   // 112
  BO_AB1  = BO_AB0  + 112,
  BO_M3B0 = BO_AB1  + 112,  // 160
  BO_M3B1 = BO_M3B0 + 160,
  BO_M3B2 = BO_M3B1 + 112,
  BO_END  = BO_M3B2 + 112   // 1056 floats
};

// ---------------------------------------------------------------------------
// Prep: convert f32 weight (K x N, row-major) into WMMA-B-operand-native f16
// tiles: tile (ct,ks) holds 512 halves; lane l gets 16 consecutive halves =
// column n = ct*16+(l&15), k = ks*32 + (l<16 ? 0 : 8) + (j<8 ? j : j+8), zero-padded.
// ---------------------------------------------------------------------------
__global__ void pack_weight_kernel(const float* __restrict__ W, _Float16* __restrict__ dst,
                                   int K, int N, int koff, int Nsrc, int KT, int total) {
  for (int i = blockIdx.x * blockDim.x + threadIdx.x; i < total;
       i += gridDim.x * blockDim.x) {
    int j    = i & 15;
    int lane = (i >> 4) & 31;
    int tile = i >> 9;
    int ct = tile / KT, ks = tile % KT;
    int n = ct * 16 + (lane & 15);
    int k = ks * 32 + ((lane & 16) ? 8 : 0) + (j < 8 ? j : j + 8);
    float v = (k < K && n < N) ? W[(size_t)(k + koff) * Nsrc + n] : 0.0f;
    dst[i] = (_Float16)v;
  }
}

// ---------------------------------------------------------------------------
// WMMA helpers
// ---------------------------------------------------------------------------
__device__ __forceinline__ v16h load_A(const _Float16* A, int ldA, int rt, int ks, int lane) {
  int m  = lane & 15;
  int kb = ks * 32 + ((lane & 16) ? 8 : 0);           // ISA 16-bit A layout
  const _Float16* q = A + (rt * 16 + m) * ldA + kb;   // K kb..kb+7 and kb+16..kb+23
  union { uint4 u[2]; v16h h; } r;
  r.u[0] = *(const uint4*)(q);
  r.u[1] = *(const uint4*)(q + 16);
  return r.h;
}

__device__ __forceinline__ v16h load_B(const _Float16* Wp, int ct, int KT, int ks, int lane) {
  const _Float16* q = Wp + (((ct * KT + ks) << 9) + (lane << 4)); // pre-swizzled
  union { uint4 u[2]; v16h h; } r;
  r.u[0] = *(const uint4*)(q);
  r.u[1] = *(const uint4*)(q + 8);
  return r.h;
}

__device__ __forceinline__ v8f gemm_tile(const _Float16* A, int ldA, int rt, int ct,
                                         const _Float16* Wp, int KT, int lane) {
  v8f acc = {0.f, 0.f, 0.f, 0.f, 0.f, 0.f, 0.f, 0.f};
  for (int ks = 0; ks < KT; ++ks) {
    v16h a = load_A(A, ldA, rt, ks, lane);
    v16h b = load_B(Wp, ct, KT, ks, lane);
    acc = __builtin_amdgcn_wmma_f32_16x16x32_f16(false, a, false, b, (short)0, acc,
                                                 false, false);
  }
  return acc;
}

// D layout: lane<16 -> N=lane,   M=rt*16+r ; lane>=16 -> N=lane-16, M=rt*16+8+r
// biasLds is zero-padded, so pad columns stay exactly 0 without predication.
__device__ __forceinline__ void store_tile(_Float16* D, int ldD, int rt, int ct, v8f acc,
                                           const float* biasLds, bool relu, int lane) {
  int n  = ct * 16 + (lane & 15);
  float bv = biasLds[n];
  int mb = rt * 16 + ((lane & 16) ? 8 : 0);
#pragma unroll
  for (int r = 0; r < 8; ++r) {
    float v = acc[r] + bv;
    if (relu) v = fmaxf(v, 0.0f);
    D[(mb + r) * ldD + n] = (_Float16)v;
  }
}

#define FOR_TILES(RTn, CTn) for (int t = wid; t < (RTn) * (CTn); t += WAVES)

// ---------------------------------------------------------------------------
// Fused value network: one block = 8 batch elements, everything in LDS.
// ---------------------------------------------------------------------------
__global__ void __launch_bounds__(256)
valuenet_kernel(const float* __restrict__ state, const _Float16* __restrict__ wp,
                const float* __restrict__ m1b0, const float* __restrict__ m1b1,
                const float* __restrict__ m2b0, const float* __restrict__ m2b1,
                const float* __restrict__ ab0,  const float* __restrict__ ab1,
                const float* __restrict__ aw2,  const float* __restrict__ ab2,
                const float* __restrict__ m3b0, const float* __restrict__ m3b1,
                const float* __restrict__ m3b2, const float* __restrict__ m3w3,
                const float* __restrict__ m3b3, float* __restrict__ out) {
  __shared__ _Float16 sBufA[RR * 160];   // h1 -> f1 -> a1 (reused)
  __shared__ _Float16 sH   [RR * 128];   // h  -> a2 (reused)
  __shared__ _Float16 sF   [RR * 64];    // f
  __shared__ _Float16 sX   [RR * 32];    // padded input
  __shared__ _Float16 sGh  [16 * 128];   // mean(h) as WMMA A (rows 0..7)
  __shared__ float    sGp  [BE * 112];   // g @ aw0_hi + ab0 (per-element bias)
  __shared__ float    sA3  [RR];         // attention logits
  __shared__ _Float16 sJoint[16 * 64];   // [self_state(6), weighted(50)] padded
  __shared__ _Float16 sV1  [16 * 160];
  __shared__ _Float16 sV2  [16 * 128];
  __shared__ _Float16 sV3  [16 * 128];
  __shared__ float    sB   [BO_END];     // zero-padded biases

  const int tid  = threadIdx.x;
  const int wid  = tid >> 5;
  const int lane = tid & 31;
  const int be0  = blockIdx.x * BE;

  // ---- stage 0: load/pad input, stage biases, zero small buffers ----------
  for (int i = tid; i < RR * 32; i += 256) {
    int r = i >> 5, k = i & 31;
    int e = r / 20, nd = r % 20;
    float v = (k < 13) ? state[((size_t)(be0 + e) * 20 + nd) * 13 + k] : 0.0f;
    sX[i] = (_Float16)v;
  }
  {
    auto ldb = [&](const float* g, int off, int nvalid, int npad) {
      for (int i = tid; i < npad; i += 256) sB[off + i] = (i < nvalid) ? g[i] : 0.0f;
    };
    ldb(m1b0, BO_M1B0, 150, 160);
    ldb(m1b1, BO_M1B1, 100, 112);
    ldb(m2b0, BO_M2B0, 100, 112);
    ldb(m2b1, BO_M2B1, 50,  64);
    ldb(ab0,  BO_AB0,  100, 112);
    ldb(ab1,  BO_AB1,  100, 112);
    ldb(m3b0, BO_M3B0, 150, 160);
    ldb(m3b1, BO_M3B1, 100, 112);
    ldb(m3b2, BO_M3B2, 100, 112);
  }
  for (int i = tid; i < 16 * 64; i += 256)  sJoint[i] = (_Float16)0.0f;
  for (int i = tid; i < 16 * 128; i += 256) sGh[i]    = (_Float16)0.0f;
  // Prefetch next block's input slab while we compute (global_prefetch_b8).
  if (blockIdx.x + 1 < gridDim.x) {
    const char* nxt = (const char*)(state + (size_t)(be0 + BE) * 20 * 13);
    for (int i = tid; i < (RR * 13 * 4) / 128; i += 256)
      __builtin_prefetch(nxt + i * 128, 0, 1);
  }
  __syncthreads();
  if (tid < BE * 6) {  // self_state = state[:,0,:6]
    int e = tid / 6, c = tid % 6;
    sJoint[e * 64 + c] = (_Float16)state[((size_t)(be0 + e) * 20) * 13 + c];
  }

  // ---- h1 = relu(X @ m1w0 + b) : 160x13 -> 160x150 ------------------------
  FOR_TILES(RT, 10) {
    int rt = t / 10, ct = t % 10;
    v8f acc = gemm_tile(sX, 32, rt, ct, wp + OW_M1W0, 1, lane);
    store_tile(sBufA, 160, rt, ct, acc, sB + BO_M1B0, true, lane);
  }
  __syncthreads();

  // ---- h = relu(h1 @ m1w1 + b) : 150 -> 100 -------------------------------
  FOR_TILES(RT, 7) {
    int rt = t / 7, ct = t % 7;
    v8f acc = gemm_tile(sBufA, 160, rt, ct, wp + OW_M1W1, 5, lane);
    store_tile(sH, 128, rt, ct, acc, sB + BO_M1B1, true, lane);
  }
  __syncthreads();

  // ---- g = mean(h over nodes), as f16 WMMA-A rows 0..7 --------------------
  for (int i = tid; i < BE * 100; i += 256) {
    int e = i / 100, c = i % 100;
    float s = 0.0f;
    for (int nd = 0; nd < 20; ++nd) s += (float)sH[(e * 20 + nd) * 128 + c];
    sGh[e * 128 + c] = (_Float16)(s * 0.05f);
  }
  __syncthreads();

  // ---- gpart = g @ aw0[100:200] + ab0 (per-element attention bias) --------
  FOR_TILES(1, 7) {
    int ct = t;
    v8f acc = gemm_tile(sGh, 128, 0, ct, wp + OW_AW0HI, 4, lane);
    int n = ct * 16 + (lane & 15);
    float bv = sB[BO_AB0 + n];
    int mb = (lane & 16) ? 8 : 0;
#pragma unroll
    for (int r = 0; r < 8; ++r) {
      int m = mb + r;
      if (m < BE) sGp[m * 112 + n] = acc[r] + bv;
    }
  }
  __syncthreads();

  // ---- f1 = relu(h @ m2w0 + b) --------------------------------------------
  FOR_TILES(RT, 7) {
    int rt = t / 7, ct = t % 7;
    v8f acc = gemm_tile(sH, 128, rt, ct, wp + OW_M2W0, 4, lane);
    store_tile(sBufA, 128, rt, ct, acc, sB + BO_M2B0, true, lane);
  }
  __syncthreads();

  // ---- f = f1 @ m2w1 + b (no relu) : -> 160x50 ----------------------------
  FOR_TILES(RT, 4) {
    int rt = t / 4, ct = t % 4;
    v8f acc = gemm_tile(sBufA, 128, rt, ct, wp + OW_M2W1, 4, lane);
    store_tile(sF, 64, rt, ct, acc, sB + BO_M2B1, false, lane);
  }
  __syncthreads();

  // ---- a1 = relu(h @ aw0[:100] + gpart) -----------------------------------
  FOR_TILES(RT, 7) {
    int rt = t / 7, ct = t % 7;
    v8f acc = gemm_tile(sH, 128, rt, ct, wp + OW_AW0LO, 4, lane);
    int n  = ct * 16 + (lane & 15);
    int mb = rt * 16 + ((lane & 16) ? 8 : 0);
#pragma unroll
    for (int r = 0; r < 8; ++r) {
      int m = mb + r;
      float v = fmaxf(acc[r] + sGp[(m / 20) * 112 + n], 0.0f);
      sBufA[m * 128 + n] = (_Float16)v;
    }
  }
  __syncthreads();

  // ---- a2 = relu(a1 @ aw1 + b) --------------------------------------------
  FOR_TILES(RT, 7) {
    int rt = t / 7, ct = t % 7;
    v8f acc = gemm_tile(sBufA, 128, rt, ct, wp + OW_AW1, 4, lane);
    store_tile(sH, 128, rt, ct, acc, sB + BO_AB1, true, lane);
  }
  __syncthreads();

  // ---- a3 = a2 @ aw2 + ab2 (matvec, VALU) ---------------------------------
  for (int r = tid; r < RR; r += 256) {
    float s = ab2[0];
    for (int c = 0; c < 100; ++c) s += (float)sH[r * 128 + c] * aw2[c];
    sA3[r] = s;
  }
  __syncthreads();

  // ---- weighted = sum_nodes f * a3 ; joint = [self, weighted] -------------
  for (int i = tid; i < BE * 50; i += 256) {
    int e = i / 50, c = i % 50;
    float s = 0.0f;
    for (int nd = 0; nd < 20; ++nd)
      s += (float)sF[(e * 20 + nd) * 64 + c] * sA3[e * 20 + nd];
    sJoint[e * 64 + 6 + c] = (_Float16)s;
  }
  __syncthreads();

  // ---- value head: 56 -> 150 -> 100 -> 100 -> 1 ---------------------------
  FOR_TILES(1, 10) {
    v8f acc = gemm_tile(sJoint, 64, 0, t, wp + OW_M3W0, 2, lane);
    store_tile(sV1, 160, 0, t, acc, sB + BO_M3B0, true, lane);
  }
  __syncthreads();
  FOR_TILES(1, 7) {
    v8f acc = gemm_tile(sV1, 160, 0, t, wp + OW_M3W1, 5, lane);
    store_tile(sV2, 128, 0, t, acc, sB + BO_M3B1, true, lane);
  }
  __syncthreads();
  FOR_TILES(1, 7) {
    v8f acc = gemm_tile(sV2, 128, 0, t, wp + OW_M3W2, 4, lane);
    store_tile(sV3, 128, 0, t, acc, sB + BO_M3B2, true, lane);
  }
  __syncthreads();

  if (tid < BE) {
    float s = m3b3[0];
    for (int c = 0; c < 100; ++c) s += (float)sV3[tid * 128 + c] * m3w3[c];
    out[be0 + tid] = s;
  }
}

// ---------------------------------------------------------------------------
extern "C" void kernel_launch(void* const* d_in, const int* in_sizes, int n_in,
                              void* d_out, int out_size, void* d_ws, size_t ws_size,
                              hipStream_t stream) {
  const float* state = (const float*)d_in[0];
  const float* m1w0 = (const float*)d_in[1];  const float* m1b0 = (const float*)d_in[2];
  const float* m1w1 = (const float*)d_in[3];  const float* m1b1 = (const float*)d_in[4];
  const float* m2w0 = (const float*)d_in[5];  const float* m2b0 = (const float*)d_in[6];
  const float* m2w1 = (const float*)d_in[7];  const float* m2b1 = (const float*)d_in[8];
  const float* aw0  = (const float*)d_in[9];  const float* ab0  = (const float*)d_in[10];
  const float* aw1  = (const float*)d_in[11]; const float* ab1  = (const float*)d_in[12];
  const float* aw2  = (const float*)d_in[13]; const float* ab2  = (const float*)d_in[14];
  const float* m3w0 = (const float*)d_in[15]; const float* m3b0 = (const float*)d_in[16];
  const float* m3w1 = (const float*)d_in[17]; const float* m3b1 = (const float*)d_in[18];
  const float* m3w2 = (const float*)d_in[19]; const float* m3b2 = (const float*)d_in[20];
  const float* m3w3 = (const float*)d_in[21]; const float* m3b3 = (const float*)d_in[22];
  _Float16* wp = (_Float16*)d_ws;

  auto prep = [&](const float* W, int K, int N, int koff, int Nsrc, int KT, int NT, int off) {
    int total = NT * KT * 512;
    pack_weight_kernel<<<(total + 255) / 256, 256, 0, stream>>>(
        W, wp + off, K, N, koff, Nsrc, KT, total);
  };
  prep(m1w0, 13,  150, 0,   150, 1, 10, OW_M1W0);
  prep(m1w1, 150, 100, 0,   100, 5, 7,  OW_M1W1);
  prep(m2w0, 100, 100, 0,   100, 4, 7,  OW_M2W0);
  prep(m2w1, 100, 50,  0,   50,  4, 4,  OW_M2W1);
  prep(aw0,  100, 100, 0,   100, 4, 7,  OW_AW0LO);
  prep(aw0,  100, 100, 100, 100, 4, 7,  OW_AW0HI);
  prep(aw1,  100, 100, 0,   100, 4, 7,  OW_AW1);
  prep(m3w0, 56,  150, 0,   150, 2, 10, OW_M3W0);
  prep(m3w1, 150, 100, 0,   100, 5, 7,  OW_M3W1);
  prep(m3w2, 100, 100, 0,   100, 4, 7,  OW_M3W2);

  const int nblocks = 16384 / BE;  // 2048
  valuenet_kernel<<<nblocks, 256, 0, stream>>>(
      state, wp, m1b0, m1b1, m2b0, m2b1, ab0, ab1, aw2, ab2,
      m3b0, m3b1, m3b2, m3w3, m3b3, (float*)d_out);
}
